// MultiHeadAttention_57054345560512
// MI455X (gfx1250) — compile-verified, matrix operand order checked by reading, not run on hardware
//
#include <hip/hip_runtime.h>

// ---------------------------------------------------------------------------
// MHA forward for MI455X (gfx1250): f16 WMMA GEMMs + flash attention.
// B=4, S=2048, D=1024, H=16, HD=64
// ---------------------------------------------------------------------------

typedef __attribute__((ext_vector_type(16))) _Float16 v16h;
typedef __attribute__((ext_vector_type(8)))  _Float16 v8h;
typedef __attribute__((ext_vector_type(8)))  float    v8f;

#define B_  4
#define S_  2048
#define D_  1024
#define H_  16
#define HD_ 64

// ---------------------------------------------------------------------------
// Fragment loader: 16x32 f16 tile, row-major source (row stride in halves).
// Matches CDNA5 16-bit A-matrix layout (ISA 7.12.2):
//   lanes 0-15 : row M = lane,    halves 0-7 -> K=0..7,  halves 8-15 -> K=16..23
//   lanes 16-31: row M = lane-16, halves 0-7 -> K=8..15, halves 8-15 -> K=24..31
// Because y = x @ W^T, row-major W *is* the B matrix in [n][k] order, so the
// same loader serves A and B fragments. Two 16B ds_load_b128 per lane.
// ---------------------------------------------------------------------------
__device__ __forceinline__ v16h load_frag(const _Float16* base, int stride) {
  const int lane = threadIdx.x & 31;
  const int r    = lane & 15;
  const int kg   = lane >> 4;          // 0 or 1
  const _Float16* p = base + r * stride + kg * 8;
  union { v16h v; v8h h[2]; } u;
  u.h[0] = *(const v8h*)(p);           // K = kg*8 .. kg*8+7
  u.h[1] = *(const v8h*)(p + 16);      // K = 16+kg*8 .. 16+kg*8+7
  return u.v;
}

__device__ __forceinline__ v8f wmma16(v16h a, v16h b, v8f c) {
  // D = A(16x32 f16) * B(32x16 f16) + C(16x16 f32)
  return __builtin_amdgcn_wmma_f32_16x16x32_f16(false, a, false, b,
                                                (short)0, c, false, false);
}

// ---------------------------------------------------------------------------
// Row reductions across a 16-lane group via DPP16 ROW_XMASK (pure VALU,
// no LDS/ds_bpermute traffic). ROW_XMASK ctrl = 0x160 | mask: lane <-> lane^mask
// within each 16-lane row -- exactly the butterfly we need on wave32.
// ---------------------------------------------------------------------------
template <int CTRL>
__device__ __forceinline__ float dpp_movf(float x) {
  return __int_as_float(__builtin_amdgcn_update_dpp(
      0, __float_as_int(x), CTRL, 0xf, 0xf, true));
}
__device__ __forceinline__ float redmax16(float v) {
  v = fmaxf(v, dpp_movf<0x161>(v));   // xor 1
  v = fmaxf(v, dpp_movf<0x162>(v));   // xor 2
  v = fmaxf(v, dpp_movf<0x164>(v));   // xor 4
  v = fmaxf(v, dpp_movf<0x168>(v));   // xor 8
  return v;
}
__device__ __forceinline__ float redsum16(float v) {
  v += dpp_movf<0x161>(v);
  v += dpp_movf<0x162>(v);
  v += dpp_movf<0x164>(v);
  v += dpp_movf<0x168>(v);
  return v;
}

// ---------------------------------------------------------------------------
// f32 -> f16 conversion (query and weights), 4 elements/thread.
// ---------------------------------------------------------------------------
__global__ void cvt_f32_to_f16(const float* __restrict__ in,
                               _Float16* __restrict__ out, int n) {
  int i = (blockIdx.x * blockDim.x + threadIdx.x) * 4;
  if (i + 3 < n) {
    float4 v = *(const float4*)(in + i);
    out[i + 0] = (_Float16)v.x;
    out[i + 1] = (_Float16)v.y;
    out[i + 2] = (_Float16)v.z;
    out[i + 3] = (_Float16)v.w;
  }
}

// ---------------------------------------------------------------------------
// GEMM: y[m][n] = sum_k A[m][k] * W[n][k] + bias[n]
// Block tile 128(M) x 128(N), TK = 32, double-buffered LDS (ping-pong,
// one barrier per K step), register prefetch of the next K slab.
// 128 threads = 4 waves, each wave owns a 64x64 output tile (16 acc tiles,
// 16 wmma per K step, 1:1 ds_load_b128 : wmma ratio).
// MODE=1: f16 head-major [B,H,S,HD] output, scaled (QKV path).
// MODE=0: f32 row-major [M,N] output (final projection).
// ---------------------------------------------------------------------------
template <int MODE>
__global__ void __launch_bounds__(128)
gemm_wmma(const _Float16* __restrict__ A,   // [8192][1024] f16
          const _Float16* __restrict__ W,   // [1024][1024] f16 (row-major)
          const float*    __restrict__ bias,
          _Float16* __restrict__ outH, float* __restrict__ outF,
          float scale) {
  __shared__ _Float16 As[2][128 * 40];   // 128 rows x 32 K, stride 40 (pad)
  __shared__ _Float16 Bs[2][128 * 40];

  const int t    = threadIdx.x;          // 0..127
  const int lane = t & 31;
  const int wid  = t >> 5;               // 0..3
  const int wm   = wid & 1;              // M sub-tile (x64)
  const int wn   = wid >> 1;             // N sub-tile (x64)
  const int m0   = blockIdx.y * 128;
  const int n0   = blockIdx.x * 128;
  const int cl   = lane & 15;
  const int rh   = (lane >> 4) * 8;

  v8f acc[4][4];
#pragma unroll
  for (int i = 0; i < 4; ++i)
#pragma unroll
    for (int j = 0; j < 4; ++j) acc[i][j] = {};

  // Cooperative stage: thread t owns one full 32-half row slab per matrix.
  const _Float16* ga = A + (size_t)(m0 + t) * D_;
  const _Float16* gb = W + (size_t)(n0 + t) * D_;

  v8h ra[4], rb[4];
#pragma unroll
  for (int u = 0; u < 4; ++u) {
    ra[u] = *(const v8h*)(ga + u * 8);
    rb[u] = *(const v8h*)(gb + u * 8);
  }
#pragma unroll
  for (int u = 0; u < 4; ++u) {
    *(v8h*)&As[0][t * 40 + u * 8] = ra[u];
    *(v8h*)&Bs[0][t * 40 + u * 8] = rb[u];
  }
  __syncthreads();

  const int NIT = D_ / 32;               // 32
  for (int it = 0; it < NIT; ++it) {
    const int buf = it & 1;
    const bool last = (it == NIT - 1);
    if (!last) {
      const int k1 = (it + 1) * 32;
#pragma unroll
      for (int u = 0; u < 4; ++u) {
        ra[u] = *(const v8h*)(ga + k1 + u * 8);
        rb[u] = *(const v8h*)(gb + k1 + u * 8);
      }
    }

    v16h af[4];
#pragma unroll
    for (int i = 0; i < 4; ++i)
      af[i] = load_frag(&As[buf][(wm * 64 + i * 16) * 40], 40);
#pragma unroll
    for (int j = 0; j < 4; ++j) {
      v16h bf = load_frag(&Bs[buf][(wn * 64 + j * 16) * 40], 40);
#pragma unroll
      for (int i = 0; i < 4; ++i)
        acc[i][j] = wmma16(af[i], bf, acc[i][j]);
    }

    if (!last) {
      const int nb = buf ^ 1;
#pragma unroll
      for (int u = 0; u < 4; ++u) {
        *(v8h*)&As[nb][t * 40 + u * 8] = ra[u];
        *(v8h*)&Bs[nb][t * 40 + u * 8] = rb[u];
      }
      __syncthreads();
    }
  }

  // Epilogue. bias depends only on (j, cl): 4 loads per thread.
  const int nwave = n0 + wn * 64;
  float bj[4];
#pragma unroll
  for (int j = 0; j < 4; ++j) bj[j] = bias[nwave + j * 16 + cl];

  if (MODE == 1) {
    // m0 is 128-aligned -> batch index constant per block; nwave is
    // 64-aligned -> head index constant per wave.
    const int b  = m0 >> 11;
    const int sb = (m0 & (S_ - 1)) + wm * 64;
    const int h  = nwave >> 6;
    _Float16* ph = outH + ((size_t)b * H_ + h) * S_ * HD_;
#pragma unroll
    for (int i = 0; i < 4; ++i) {
#pragma unroll
      for (int r = 0; r < 8; ++r) {
        _Float16* pr = ph + (size_t)(sb + i * 16 + rh + r) * HD_;
#pragma unroll
        for (int j = 0; j < 4; ++j)
          pr[j * 16 + cl] = (_Float16)((acc[i][j][r] + bj[j]) * scale);
      }
    }
  } else {
    float* pm = outF + (size_t)(m0 + wm * 64) * D_ + nwave;
#pragma unroll
    for (int i = 0; i < 4; ++i) {
#pragma unroll
      for (int r = 0; r < 8; ++r) {
        float* pr = pm + (size_t)(i * 16 + rh + r) * D_;
#pragma unroll
        for (int j = 0; j < 4; ++j)
          pr[j * 16 + cl] = acc[i][j][r] + bj[j];
      }
    }
  }
}

// ---------------------------------------------------------------------------
// Flash attention: block = (b, h, 128 query rows); 8 waves x 16 rows each.
// KV tile = 64 keys per iteration -> 16 WMMA per barrier pair (8 QK^T + 8 PV).
// Q lives in registers (2 A-fragments, softmax scale pre-folded). K staged
// row-major in LDS (B-fragment-ready); V staged transposed. Next tile is
// register-prefetched across the compute phase; global_prefetch_b8 two tiles
// ahead. Online softmax per row via DPP ROW_XMASK butterflies (pure VALU).
// ---------------------------------------------------------------------------
__global__ void __launch_bounds__(256)
attention_wmma(const _Float16* __restrict__ q16,   // [B,H,S,HD] (pre-scaled)
               const _Float16* __restrict__ k16,   // [B,H,S,HD]
               const _Float16* __restrict__ v16,   // [B,H,S,HD]
               _Float16* __restrict__ a16) {       // [B,S,D]
  __shared__ _Float16 Ks[64 * 72];        // [n][d]   n=0..63, stride 72 (pad)
  __shared__ _Float16 Vt[64 * 72];        // [d][n]   d=0..63, stride 72 (pad)
  __shared__ _Float16 Ps[8][16 * 72];     // per-wave P staging [16][64+pad]

  const int t    = threadIdx.x;
  const int lane = t & 31;
  const int wid  = t >> 5;
  const int cl   = lane & 15;
  const int rh   = (lane >> 4) * 8;

  const int qt = blockIdx.x;              // query tile (of 128 rows)
  const int h  = blockIdx.y;
  const int b  = blockIdx.z;

  const size_t head = (size_t)(b * H_ + h) * S_ * HD_;
  const _Float16* qb = q16 + head;
  const _Float16* kb = k16 + head;
  const _Float16* vb = v16 + head;

  const int qr0 = qt * 128 + wid * 16;    // this wave's 16 query rows

  // Q fragments: 16 rows x 64 dims -> two 16x32 fragments.
  v16h qa0 = load_frag(qb + (size_t)qr0 * HD_,      HD_);
  v16h qa1 = load_frag(qb + (size_t)qr0 * HD_ + 32, HD_);

  v8f o[4];
#pragma unroll
  for (int j = 0; j < 4; ++j) o[j] = {};
  float mrow[8], lrow[8];
#pragma unroll
  for (int r = 0; r < 8; ++r) { mrow[r] = -1e30f; lrow[r] = 0.0f; }

  _Float16* Pw = &Ps[wid][0];

  // Cooperative stage mapping: 64 rows x 64 halves, 16 halves per thread.
  const int srow = t >> 2;                // 0..63
  const int scol = (t & 3) * 16;          // 0/16/32/48

  // Preload first K/V tile into registers.
  v8h kr0 = *(const v8h*)(kb + (size_t)srow * HD_ + scol);
  v8h kr1 = *(const v8h*)(kb + (size_t)srow * HD_ + scol + 8);
  v8h vr0 = *(const v8h*)(vb + (size_t)srow * HD_ + scol);
  v8h vr1 = *(const v8h*)(vb + (size_t)srow * HD_ + scol + 8);

  for (int kv0 = 0; kv0 < S_; kv0 += 64) {
    // Commit staged registers: K row-major, V transposed.
    *(v8h*)&Ks[srow * 72 + scol]     = kr0;
    *(v8h*)&Ks[srow * 72 + scol + 8] = kr1;
#pragma unroll
    for (int i = 0; i < 8; ++i) {
      Vt[(scol + i) * 72 + srow]     = vr0[i];
      Vt[(scol + 8 + i) * 72 + srow] = vr1[i];
    }
    __syncthreads();

    // Prefetch next tile into registers (hidden behind the WMMAs below).
    if (kv0 + 64 < S_) {
      const _Float16* kp = kb + (size_t)(kv0 + 64 + srow) * HD_ + scol;
      const _Float16* vp = vb + (size_t)(kv0 + 64 + srow) * HD_ + scol;
      kr0 = *(const v8h*)(kp);
      kr1 = *(const v8h*)(kp + 8);
      vr0 = *(const v8h*)(vp);
      vr1 = *(const v8h*)(vp + 8);
    }
    if (kv0 + 128 < S_) {
      __builtin_prefetch(kb + (size_t)(kv0 + 128 + srow) * HD_ + scol, 0, 3);
      __builtin_prefetch(vb + (size_t)(kv0 + 128 + srow) * HD_ + scol, 0, 3);
    }

    // Scores: four 16x16 tiles (keys kv0+16*nt). Hoist all K fragments so
    // the ds_load_b128s batch ahead of the WMMA burst.
    v16h kf0l = load_frag(&Ks[0 * 16 * 72],      72);
    v16h kf0h = load_frag(&Ks[0 * 16 * 72 + 32], 72);
    v16h kf1l = load_frag(&Ks[1 * 16 * 72],      72);
    v16h kf1h = load_frag(&Ks[1 * 16 * 72 + 32], 72);
    v16h kf2l = load_frag(&Ks[2 * 16 * 72],      72);
    v16h kf2h = load_frag(&Ks[2 * 16 * 72 + 32], 72);
    v16h kf3l = load_frag(&Ks[3 * 16 * 72],      72);
    v16h kf3h = load_frag(&Ks[3 * 16 * 72 + 32], 72);

    v8f sc[4];
    sc[0] = wmma16(qa1, kf0h, wmma16(qa0, kf0l, (v8f){}));
    sc[1] = wmma16(qa1, kf1h, wmma16(qa0, kf1l, (v8f){}));
    sc[2] = wmma16(qa1, kf2h, wmma16(qa0, kf2l, (v8f){}));
    sc[3] = wmma16(qa1, kf3h, wmma16(qa0, kf3l, (v8f){}));

    // Online softmax (per row; row lives in a 16-lane group).
    float corr[8];
#pragma unroll
    for (int r = 0; r < 8; ++r) {
      float tmax = fmaxf(fmaxf(sc[0][r], sc[1][r]), fmaxf(sc[2][r], sc[3][r]));
      tmax = redmax16(tmax);
      float mnew = fmaxf(mrow[r], tmax);
      corr[r]  = __expf(mrow[r] - mnew);
      float p0 = __expf(sc[0][r] - mnew);
      float p1 = __expf(sc[1][r] - mnew);
      float p2 = __expf(sc[2][r] - mnew);
      float p3 = __expf(sc[3][r] - mnew);
      lrow[r]  = lrow[r] * corr[r] + redsum16((p0 + p1) + (p2 + p3));
      mrow[r]  = mnew;
      Pw[(rh + r) * 72 + cl]      = (_Float16)p0;
      Pw[(rh + r) * 72 + 16 + cl] = (_Float16)p1;
      Pw[(rh + r) * 72 + 32 + cl] = (_Float16)p2;
      Pw[(rh + r) * 72 + 48 + cl] = (_Float16)p3;
    }
#pragma unroll
    for (int j = 0; j < 4; ++j)
#pragma unroll
      for (int r = 0; r < 8; ++r) o[j][r] *= corr[r];

    // PV: P (16x64) x V (64x64) accumulated into 4 output tiles.
    v16h pa0 = load_frag(Pw, 72);          // keys kv0..kv0+31
    v16h pa1 = load_frag(Pw + 32, 72);     // keys kv0+32..kv0+63
#pragma unroll
    for (int j = 0; j < 4; ++j) {
      v16h vf0 = load_frag(&Vt[(j * 16) * 72],      72);
      v16h vf1 = load_frag(&Vt[(j * 16) * 72 + 32], 72);
      o[j] = wmma16(pa0, vf0, o[j]);
      o[j] = wmma16(pa1, vf1, o[j]);
    }
    __syncthreads();   // protect Ks/Vt before next stage
  }

  // Normalize and write token-major [B,S,D] (f16) for the output projection.
#pragma unroll
  for (int r = 0; r < 8; ++r) {
    const float inv = 1.0f / lrow[r];
    const int row = qr0 + rh + r;
#pragma unroll
    for (int j = 0; j < 4; ++j) {
      const int d = h * HD_ + j * 16 + cl;
      a16[((size_t)b * S_ + row) * D_ + d] = (_Float16)(o[j][r] * inv);
    }
  }
}

// ---------------------------------------------------------------------------
// Launcher
// ---------------------------------------------------------------------------
extern "C" void kernel_launch(void* const* d_in, const int* in_sizes, int n_in,
                              void* d_out, int out_size, void* d_ws, size_t ws_size,
                              hipStream_t stream) {
  (void)in_sizes; (void)n_in; (void)out_size; (void)ws_size;

  const float* query = (const float*)d_in[0];
  const float* Wq = (const float*)d_in[1]; const float* bq = (const float*)d_in[2];
  const float* Wk = (const float*)d_in[3]; const float* bk = (const float*)d_in[4];
  const float* Wv = (const float*)d_in[5]; const float* bv = (const float*)d_in[6];
  const float* Wo = (const float*)d_in[7]; const float* bo = (const float*)d_in[8];
  float* out = (float*)d_out;

  const size_t DD  = (size_t)D_ * D_;        // 1M
  const size_t BSD = (size_t)B_ * S_ * D_;   // 8M

  _Float16* w16  = (_Float16*)d_ws;          // 4*DD halves
  _Float16* wq16 = w16;
  _Float16* wk16 = w16 + DD;
  _Float16* wv16 = w16 + 2 * DD;
  _Float16* wo16 = w16 + 3 * DD;
  _Float16* x16  = w16 + 4 * DD;             // BSD
  _Float16* q16  = x16 + BSD;                // BSD  [B,H,S,HD]
  _Float16* k16  = q16 + BSD;
  _Float16* v16p = k16 + BSD;
  _Float16* a16  = v16p + BSD;               // BSD  [B,S,D]

  // 1) f32 -> f16 conversions.
  cvt_f32_to_f16<<<(int)(BSD / 4 / 256), 256, 0, stream>>>(query, x16, (int)BSD);
  cvt_f32_to_f16<<<(int)(DD  / 4 / 256), 256, 0, stream>>>(Wq, wq16, (int)DD);
  cvt_f32_to_f16<<<(int)(DD  / 4 / 256), 256, 0, stream>>>(Wk, wk16, (int)DD);
  cvt_f32_to_f16<<<(int)(DD  / 4 / 256), 256, 0, stream>>>(Wv, wv16, (int)DD);
  cvt_f32_to_f16<<<(int)(DD  / 4 / 256), 256, 0, stream>>>(Wo, wo16, (int)DD);

  // 2) QKV projections (head-major f16; softmax scale folded into Q).
  dim3 ggrid(D_ / 128, (B_ * S_) / 128);     // (8, 64)
  gemm_wmma<1><<<ggrid, 128, 0, stream>>>(x16, wq16, bq, q16, nullptr, 0.125f);
  gemm_wmma<1><<<ggrid, 128, 0, stream>>>(x16, wk16, bk, k16, nullptr, 1.0f);
  gemm_wmma<1><<<ggrid, 128, 0, stream>>>(x16, wv16, bv, v16p, nullptr, 1.0f);

  // 3) Flash attention.
  dim3 agrid(S_ / 128, H_, B_);              // (16, 16, 4)
  attention_wmma<<<agrid, 256, 0, stream>>>(q16, k16, v16p, a16);

  // 4) Output projection -> f32 d_out.
  gemm_wmma<0><<<ggrid, 128, 0, stream>>>(a16, wo16, bo, nullptr, out, 1.0f);
}
